// ImplicitNetworkGrid_45672682226376
// MI455X (gfx1250) — matrix operand order repaired and checked by reading, not compiled, not run on hardware
//
#include <hip/hip_runtime.h>

// ---------------------------------------------------------------------------
// Fused Instant-NGP hash-grid + weight-norm MLP for MI455X (gfx1250).
//   phase 0: multilevel hash encode -> f16 A matrix in LDS
//   phase 1: [128,64] x [64,256]  WMMA f16 -> softplus100 -> LDS f16
//   phase 2: [128,256] x [256,224] WMMA f16 -> softplus100/sqrt2 -> LDS f16
//            (cols 221..255 of the layer-2 input hold inp/sqrt2, the skip)
//   phase 3: [128,256] x [256,272] WMMA f16 -> +bias -> f32 global store
// Weights are normalized + swizzled into WMMA B-fragment order by prep
// kernels (output reused by all 2048 blocks; ~290 KB, L2 resident).
// A-fragments are hoisted into registers once per layer (K-step invariant
// across N-tiles); softplus uses hw v_exp/v_log (branchless).
// ---------------------------------------------------------------------------

typedef _Float16 half_t;
typedef __attribute__((ext_vector_type(16))) _Float16 v16h;
typedef __attribute__((ext_vector_type(8)))  _Float16 v8h;
typedef __attribute__((ext_vector_type(8)))  float    v8f;

#define NPTS      262144
#define TBL       524288          // 2^19 entries per level
#define TMASK     (TBL - 1)
#define ROWS_BLK  128
#define THREADS   256
#define INV_DIV   (1.0f / 1.5f)
#define INV_SQRT2 0.70710678118654752f
#define LOGB      0.32346868426130783f   // ln(2048/16)/15

// fragment-blob counts: blob = kstep * nTiles + tile ; 512 halves per blob
#define T0 16
#define K0 2      // K = 64
#define T1 14     // N = 224 (221 padded)
#define K1 8      // K = 256
#define T2 17     // N = 272 (257 padded)
#define K2 8
#define BLOBS0 (T0 * K0)                 // 32
#define BLOBS1 (T1 * K1)                 // 112
#define BLOBS2 (T2 * K2)                 // 136
#define WF0_OFF 0
#define WF1_OFF (BLOBS0 * 512)
#define WF2_OFF ((BLOBS0 + BLOBS1) * 512)
#define WF_HALVES ((BLOBS0 + BLOBS1 + BLOBS2) * 512)   // 143360

// softplus(100v)/100 with hardware exp/log; branchless (select), EXEC uniform
__device__ __forceinline__ float softplus100(float v) {
    const float t = 100.0f * v;
    const float e = __expf(fminf(t, 20.0f));       // clamp avoids inf
    const float sp = __logf(1.0f + e) * 0.01f;
    return (t > 20.0f) ? v : sp;                    // softplus(t>20) ~ t
}

// A-fragment from row-major f16 LDS (16-bit A 16x32 layout, ISA 7.12.2)
__device__ __forceinline__ v16h load_a(const half_t* act, int ld, int row,
                                       int k0, int kbase) {
    union { v16h v; v8h h[2]; } u;
    u.h[0] = *(const v8h*)(act + row * ld + k0 + kbase);
    u.h[1] = *(const v8h*)(act + row * ld + k0 + 16 + kbase);
    return u.v;
}

// B-fragment from pre-swizzled global blob (lane-major, 16 halves per lane)
__device__ __forceinline__ v16h load_b(const half_t* __restrict__ wf,
                                       int blob, int lane) {
    return *(const v16h*)(wf + blob * 512 + lane * 16);
}

__global__ __launch_bounds__(THREADS)
void fused_ngp_mlp(const float* __restrict__ x,
                   const float* __restrict__ table,
                   const half_t* __restrict__ wf,
                   const float* __restrict__ biases,
                   float* __restrict__ out) {
    extern __shared__ char smem_raw[];
    half_t* A0   = (half_t*)smem_raw;              // [128][64]
    half_t* actA = A0 + ROWS_BLK * 64;             // [128][256]
    half_t* actB = actA + ROWS_BLK * 256;          // [128][256]

    const int tid   = threadIdx.x;
    const int gRow0 = blockIdx.x * ROWS_BLK;

    // ---------------- phase 0: hash encode (2 threads / point) -------------
    {
        const int row  = tid >> 1;
        const int part = tid & 1;
        const float x0 = x[(gRow0 + row) * 3 + 0];
        const float x1 = x[(gRow0 + row) * 3 + 1];
        const float x2 = x[(gRow0 + row) * 3 + 2];
        const float u0 = (x0 * INV_DIV + 1.0f) * 0.5f;
        const float u1 = (x1 * INV_DIV + 1.0f) * 0.5f;
        const float u2 = (x2 * INV_DIV + 1.0f) * 0.5f;

        if (part == 0) {
            A0[row * 64 + 0] = (half_t)x0;
            A0[row * 64 + 1] = (half_t)x1;
            A0[row * 64 + 2] = (half_t)x2;
            actB[row * 256 + 221 + 0] = (half_t)(x0 * INV_SQRT2);
            actB[row * 256 + 221 + 1] = (half_t)(x1 * INV_SQRT2);
            actB[row * 256 + 221 + 2] = (half_t)(x2 * INV_SQRT2);
        } else {
            #pragma unroll
            for (int c = 35; c < 64; ++c) A0[row * 64 + c] = (half_t)0.0f;
        }

        const float2* __restrict__ tbl2 = (const float2*)table;
        for (int l = 0; l < 8; ++l) {
            const int level  = part * 8 + l;
            const float scale = __expf(LOGB * (float)level) * 16.0f - 1.0f;
            const float p0 = u0 * scale + 0.5f;
            const float p1 = u1 * scale + 0.5f;
            const float p2 = u2 * scale + 0.5f;
            const float fl0 = floorf(p0), fl1 = floorf(p1), fl2 = floorf(p2);
            const float fr0 = p0 - fl0, fr1 = p1 - fl1, fr2 = p2 - fl2;
            const unsigned i0 = (unsigned)(int)fl0;
            const unsigned i1 = (unsigned)(int)fl1;
            const unsigned i2 = (unsigned)(int)fl2;
            float a0 = 0.0f, a1 = 0.0f;
            #pragma unroll
            for (int c = 0; c < 8; ++c) {
                const unsigned cx = (c >> 2) & 1u;
                const unsigned cy = (c >> 1) & 1u;
                const unsigned cz = c & 1u;
                const unsigned h = (i0 + cx)
                                 ^ ((i1 + cy) * 2654435761u)
                                 ^ ((i2 + cz) * 805459861u);
                const unsigned flat = (unsigned)level * TBL + (h & TMASK);
                const float2 f = tbl2[flat];
                const float w = (cx ? fr0 : 1.0f - fr0)
                              * (cy ? fr1 : 1.0f - fr1)
                              * (cz ? fr2 : 1.0f - fr2);
                a0 += w * f.x;
                a1 += w * f.y;
            }
            const int col = 3 + 2 * level;
            A0[row * 64 + col]     = (half_t)a0;
            A0[row * 64 + col + 1] = (half_t)a1;
            actB[row * 256 + 221 + col]     = (half_t)(a0 * INV_SQRT2);
            actB[row * 256 + 221 + col + 1] = (half_t)(a1 * INV_SQRT2);
        }
    }
    __syncthreads();

    // ---------------- WMMA GEMM phases -------------------------------------
    const int wave  = tid >> 5;
    const int lane  = tid & 31;
    const int m0    = wave * 16;
    const int laneM = lane & 15;
    const int hiH   = lane >> 4;
    const int kbA   = hiH * 8;        // A-fragment lane K base
    const int mBase = m0 + hiH * 8;   // C-fragment lane M base
    const float* b0 = biases;
    const float* b1 = biases + 256;
    const float* b2 = biases + 480;

    // layer 0: [128,64] x [64,256] -> softplus100 -> actA
    {
        v16h aF[K0];
        #pragma unroll
        for (int s = 0; s < K0; ++s)
            aF[s] = load_a(A0, 64, m0 + laneM, s * 32, kbA);
        for (int t = 0; t < T0; ++t) {
            v8f acc = {};
            #pragma unroll
            for (int s = 0; s < K0; ++s) {
                v16h b = load_b(wf + WF0_OFF, s * T0 + t, lane);
                acc = __builtin_amdgcn_wmma_f32_16x16x32_f16(
                    false, aF[s], false, b, (short)0, acc, false, false);
            }
            const int n = t * 16 + laneM;
            const float bn = b0[n];
            #pragma unroll
            for (int r = 0; r < 8; ++r)
                actA[(mBase + r) * 256 + n] = (half_t)softplus100(acc[r] + bn);
        }
    }
    __syncthreads();

    // layer 1: [128,256] x [256,224] -> softplus100/sqrt2 -> actB (n<221)
    {
        v16h aF[K1];
        #pragma unroll
        for (int s = 0; s < K1; ++s)
            aF[s] = load_a(actA, 256, m0 + laneM, s * 32, kbA);
        for (int t = 0; t < T1; ++t) {
            v8f acc = {};
            #pragma unroll
            for (int s = 0; s < K1; ++s) {
                v16h b = load_b(wf + WF1_OFF, s * T1 + t, lane);
                acc = __builtin_amdgcn_wmma_f32_16x16x32_f16(
                    false, aF[s], false, b, (short)0, acc, false, false);
            }
            const int n = t * 16 + laneM;
            if (n < 221) {
                const float bn = b1[n];
                #pragma unroll
                for (int r = 0; r < 8; ++r)
                    actB[(mBase + r) * 256 + n] =
                        (half_t)(softplus100(acc[r] + bn) * INV_SQRT2);
            }
        }
    }
    __syncthreads();

    // layer 2: [128,256] x [256,272] -> +bias -> f32 out (n<257)
    {
        v16h aF[K2];
        #pragma unroll
        for (int s = 0; s < K2; ++s)
            aF[s] = load_a(actB, 256, m0 + laneM, s * 32, kbA);
        for (int t = 0; t < T2; ++t) {
            v8f acc = {};
            #pragma unroll
            for (int s = 0; s < K2; ++s) {
                v16h b = load_b(wf + WF2_OFF, s * T2 + t, lane);
                acc = __builtin_amdgcn_wmma_f32_16x16x32_f16(
                    false, aF[s], false, b, (short)0, acc, false, false);
            }
            const int n = t * 16 + laneM;
            if (n < 257) {
                const float bn = b2[n];
                #pragma unroll
                for (int r = 0; r < 8; ++r)
                    out[(size_t)(gRow0 + mBase + r) * 257 + n] = acc[r] + bn;
            }
        }
    }
}

// ---------------------------------------------------------------------------
// prep: weight-norm (g*V/||V||_row), pad, and swizzle into B-fragment blobs.
// blob = kstep*nTiles + tile; within blob element (lane, j):
//   n = tile*16 + (lane&15); k = kstep*32 + (lane<16 ? 0 : 8) + (j&7) + (j>>3)*16
// ---------------------------------------------------------------------------
__global__ void prep_frags(const float* __restrict__ V,
                           const float* __restrict__ g,
                           int rows, int cols, int nTiles,
                           half_t* __restrict__ out, int nBlobs) {
    const int id = blockIdx.x * blockDim.x + threadIdx.x;
    if (id >= nBlobs * 512) return;
    const int blob = id >> 9;
    const int e    = id & 511;
    const int lane = e >> 4;
    const int j    = e & 15;
    const int s    = blob / nTiles;
    const int t    = blob - s * nTiles;
    const int n    = t * 16 + (lane & 15);
    const int k    = s * 32 + ((lane >> 4) * 8) + (j & 7) + ((j >> 3) * 16);
    float val = 0.0f;
    if (n < rows && k < cols) {
        float ss = 0.0f;
        for (int kk = 0; kk < cols; ++kk) {
            const float v = V[n * cols + kk];
            ss += v * v;
        }
        val = g[n] * V[n * cols + k] / sqrtf(ss);
    }
    out[id] = (half_t)val;
}

__global__ void prep_bias(const float* __restrict__ b, int rows, int padded,
                          float* __restrict__ out) {
    const int i = blockIdx.x * blockDim.x + threadIdx.x;
    if (i < padded) out[i] = (i < rows) ? b[i] : 0.0f;
}

// ---------------------------------------------------------------------------
extern "C" void kernel_launch(void* const* d_in, const int* in_sizes, int n_in,
                              void* d_out, int out_size, void* d_ws, size_t ws_size,
                              hipStream_t stream) {
    const float* x     = (const float*)d_in[0];
    const float* table = (const float*)d_in[1];
    const float* V0    = (const float*)d_in[2];
    const float* g0    = (const float*)d_in[3];
    const float* b0    = (const float*)d_in[4];
    const float* V1    = (const float*)d_in[5];
    const float* g1    = (const float*)d_in[6];
    const float* b1    = (const float*)d_in[7];
    const float* V2    = (const float*)d_in[8];
    const float* g2    = (const float*)d_in[9];
    const float* b2    = (const float*)d_in[10];
    float* out = (float*)d_out;

    half_t* wf     = (half_t*)d_ws;                      // 143360 halves
    float*  biases = (float*)((char*)d_ws + WF_HALVES * sizeof(half_t));

    prep_frags<<<(BLOBS0 * 512 + 255) / 256, 256, 0, stream>>>(
        V0, g0, 256, 35, T0, wf + WF0_OFF, BLOBS0);
    prep_frags<<<(BLOBS1 * 512 + 255) / 256, 256, 0, stream>>>(
        V1, g1, 221, 256, T1, wf + WF1_OFF, BLOBS1);
    prep_frags<<<(BLOBS2 * 512 + 255) / 256, 256, 0, stream>>>(
        V2, g2, 257, 256, T2, wf + WF2_OFF, BLOBS2);
    prep_bias<<<1, 256, 0, stream>>>(b0, 256, 256, biases);
    prep_bias<<<1, 256, 0, stream>>>(b1, 221, 224, biases + 256);
    prep_bias<<<2, 256, 0, stream>>>(b2, 257, 272, biases + 480);

    const int smemBytes =
        (ROWS_BLK * 64 + 2 * ROWS_BLK * 256) * (int)sizeof(half_t);  // 144 KB
    (void)hipFuncSetAttribute((const void*)fused_ngp_mlp,
                              hipFuncAttributeMaxDynamicSharedMemorySize,
                              smemBytes);
    fused_ngp_mlp<<<NPTS / ROWS_BLK, THREADS, smemBytes, stream>>>(
        x, table, wf, biases, out);
}
